// MultiHeadedAttentionRel_52647709114819
// MI455X (gfx1250) — matrix operand order repaired and checked by reading, not compiled, not run on hardware
//
#include <hip/hip_runtime.h>

typedef __attribute__((ext_vector_type(16))) _Float16 v16h;
typedef __attribute__((ext_vector_type(8)))  float    v8f;
typedef __attribute__((ext_vector_type(4)))  int      v4i;

#define NH     16
#define DK     64
#define DMODEL 1024
#define SEQ    1024
#define NBATCH 8
#define NREL   65   // 2*32+1

#if defined(__has_builtin)
#if __has_builtin(__builtin_amdgcn_global_load_async_to_lds_b128) && \
    __has_builtin(__builtin_amdgcn_s_wait_asynccnt)
#define USE_ASYNC_LDS 1
#endif
#endif
#ifndef USE_ASYNC_LDS
#define USE_ASYNC_LDS 0
#endif

// copy 8 f16 (16B) global -> LDS; async (ASYNCcnt) when available
__device__ __forceinline__ void cp16(const _Float16* g, _Float16* l) {
#if USE_ASYNC_LDS
  __builtin_amdgcn_global_load_async_to_lds_b128(
      (__attribute__((address_space(1))) v4i*)(void*)g,
      (__attribute__((address_space(3))) v4i*)(void*)l, 0, 0);
#else
  *(uint4*)l = *(const uint4*)g;
#endif
}
__device__ __forceinline__ void wait_async() {
#if USE_ASYNC_LDS
  __builtin_amdgcn_s_wait_asynccnt(0);
#endif
}

// ---------------------------------------------------------------------------
// GEMM: Y = X @ W^T + bias.  X:[M,K] (TIN), W:[N,K] f32.
// MODE 0: out f16 [B,H,S,dk]   (Q,K projections)
// MODE 2: out f16 [B,H,dk,S]   (V projection, transposed for async staging)
// MODE 1: out f32 [M,N]        (final output projection)
// Tile: M=128 (8 waves x 16 rows), N=128 (8 frags/wave), K-step 32.
// ---------------------------------------------------------------------------
template <typename TIN, int MODE>
__global__ __launch_bounds__(256) void gemm_wt(const TIN* __restrict__ X,
                                               const float* __restrict__ W,
                                               const float* __restrict__ bias,
                                               void* __restrict__ out,
                                               int M, int N, int K) {
  __shared__ _Float16 As[128][36];  // padded stride to dodge bank conflicts
  __shared__ _Float16 Bs[128][36];

  const int tid  = threadIdx.x;
  const int wave = tid >> 5;
  const int lane = tid & 31;
  const int hi   = lane >> 4;
  const int ln   = lane & 15;
  const int m0   = blockIdx.y * 128;
  const int n0   = blockIdx.x * 128;

  v8f acc[8] = {};

  for (int k0 = 0; k0 < K; k0 += 32) {
    __syncthreads();
    for (int i = tid; i < 128 * 32; i += 256) {
      int r = i >> 5, c = i & 31;
      As[r][c] = (_Float16)(float)X[(size_t)(m0 + r) * K + k0 + c];
      Bs[r][c] = (_Float16)W[(size_t)(n0 + r) * K + k0 + c];
    }
    if (k0 + 32 < K) {  // prefetch next K-panel (global_prefetch_b8)
      int r = tid >> 1;
      __builtin_prefetch(&X[(size_t)(m0 + r) * K + k0 + 32], 0, 1);
      __builtin_prefetch(&W[(size_t)(n0 + r) * K + k0 + 32], 0, 1);
    }
    __syncthreads();

    // A fragment: 16x32, rows = wave*16 .. +15
    const int mrow = wave * 16 + ln;
    union { v16h v; unsigned u[8]; } a;
#pragma unroll
    for (int q = 0; q < 4; ++q) {
      int kk = 2 * q + hi * 8;
      a.u[q]     = *(const unsigned*)&As[mrow][kk];
      a.u[q + 4] = *(const unsigned*)&As[mrow][kk + 16];
    }
#pragma unroll
    for (int f = 0; f < 8; ++f) {
      const int ncol = f * 16 + ln;
      union { v16h v; unsigned u[8]; } bf;
#pragma unroll
      for (int q = 0; q < 8; ++q) {
        int kk = 2 * (q & 3) + hi * 8 + (q >> 2) * 16;
        bf.u[q] = *(const unsigned*)&Bs[ncol][kk];
      }
      acc[f] = __builtin_amdgcn_wmma_f32_16x16x32_f16(
          false, a.v, false, bf.v, (short)0, acc[f], false, false);
    }
  }

  // Epilogue. C layout: N = lane&15 ; VGPR r -> M = r + 8*hi
#pragma unroll
  for (int f = 0; f < 8; ++f) {
#pragma unroll
    for (int r = 0; r < 8; ++r) {
      int m = m0 + wave * 16 + r + hi * 8;
      int n = n0 + f * 16 + ln;
      float val = acc[f][r] + bias[n];
      if (MODE == 0) {
        int b = m >> 10, s = m & (SEQ - 1);
        int h = n >> 6,  d = n & (DK - 1);
        ((_Float16*)out)[((size_t)(b * NH + h) * SEQ + s) * DK + d] = (_Float16)val;
      } else if (MODE == 2) {
        int b = m >> 10, s = m & (SEQ - 1);
        int h = n >> 6,  d = n & (DK - 1);
        ((_Float16*)out)[((size_t)(b * NH + h) * DK + d) * SEQ + s] = (_Float16)val;
      } else {
        ((float*)out)[(size_t)m * N + n] = val;
      }
    }
  }
}

// ---------------------------------------------------------------------------
// Attention with relative position bias.
// One block = one (bh, 32-row q-tile). 256 threads = 8 waves:
//   wq = wave>>2 selects 16 q-rows, wn = wave&3 selects 16 cols.
// Full 32x1024 f32 score strip lives in LDS (two-pass softmax).
// K/V tiles double-buffered via async global->LDS copies.
// ---------------------------------------------------------------------------
#define SM_QS  (32 * 72)       // f16 Q tile
#define SM_KV  (2 * 64 * 72)   // f16 K/V tile, double-buffered
#define SM_S   (32 * 1028)     // f32 scores
#define SM_PS  (32 * 1032)     // f16 exp(P)
#define SM_QR  (32 * 66)       // f32 qrel (reused for b64 partials)
#define SM_BK  (32 * 66)       // f32 bucket sums
#define SM_TV  (NREL * 64)     // f32 relv table
#define SM_RED (32 * 8)        // f32 reduction scratch
#define ATTN_SMEM_BYTES \
  (2 * (SM_QS + SM_KV + SM_PS) + 4 * (SM_S + SM_QR + SM_BK + SM_TV + SM_RED + 64))

__global__ __launch_bounds__(256) void attn_kernel(
    const _Float16* __restrict__ Qw, const _Float16* __restrict__ Kw,
    const _Float16* __restrict__ Vt, const float* __restrict__ relk,
    const float* __restrict__ relv, _Float16* __restrict__ ctx) {
  extern __shared__ char smem[];
  _Float16* Qs  = (_Float16*)smem;            // [32][72]
  _Float16* KV0 = Qs + SM_QS;                 // [64][72] buffer 0
  _Float16* KV1 = KV0 + 64 * 72;              // [64][72] buffer 1
  float*    S   = (float*)(KV0 + SM_KV);      // [32][1028]
  _Float16* Ps  = (_Float16*)(S + SM_S);      // [32][1032]
  float*    qrel = (float*)(Ps + SM_PS);      // [32][66]
  float*    buck = qrel + SM_QR;              // [32][66]
  float*    tv   = buck + SM_BK;              // [65][64]
  float*    red  = tv + SM_TV;                // [32][8]
  float*    rmax = red + SM_RED;              // [32]
  float*    rinv = rmax + 32;                 // [32]

  const int tid  = threadIdx.x;
  const int wave = tid >> 5;
  const int lane = tid & 31;
  const int hi   = lane >> 4;
  const int ln   = lane & 15;
  const int wq   = wave >> 2;                 // 0..1
  const int wn   = wave & 3;                  // 0..3
  const int q0   = blockIdx.x * 32;
  const int bh   = blockIdx.y;                // b*NH + h
  const size_t kbase = (size_t)bh * SEQ * DK; // K: [bh][s][d]
  const size_t vbase = (size_t)bh * DK * SEQ; // V: [bh][d][s]

  // --- stage Q tile (async) + K tile 0; relv table; zero buckets ------------
  for (int e = tid; e < 32 * 8; e += 256) {   // 32 rows x 8 chunks of 8 halfs
    int r = e >> 3, c8 = (e & 7) * 8;
    cp16(&Qw[kbase + (size_t)(q0 + r) * DK + c8], &Qs[r * 72 + c8]);
  }
  for (int e = tid; e < 64 * 8; e += 256) {   // K tile 0 -> KV0
    int r = e >> 3, c8 = (e & 7) * 8;
    cp16(&Kw[kbase + (size_t)r * DK + c8], &KV0[r * 72 + c8]);
  }
  for (int i = tid; i < NREL * 64; i += 256) tv[i] = relv[i];
  for (int i = tid; i < 32 * 66; i += 256) buck[i] = 0.f;
  wait_async();
  __syncthreads();

  // --- qrel[q][r] = Q[q,:] . relk_table[r,:] --------------------------------
  for (int e = tid; e < 32 * NREL; e += 256) {
    int q = e / NREL, r = e % NREL;
    float s = 0.f;
    for (int d = 0; d < 64; ++d) s += (float)Qs[q * 72 + d] * relk[r * 64 + d];
    qrel[q * 66 + r] = s;
  }
  __syncthreads();  // qrel visible; KV0 already complete

  const int mrow = wq * 16 + ln;

  // --- pass 1: scores = (Q K^T + qrel[idx]) / 8, double-buffered K ----------
#pragma unroll 2
  for (int kt = 0; kt < 16; ++kt) {
    const int cur = kt & 1;
    wait_async();
    __syncthreads();  // KV[cur] ready; all waves done reading KV[1-cur]
    if (kt + 1 < 16) {
      _Float16* nxt = cur ? KV0 : KV1;
      for (int e = tid; e < 64 * 8; e += 256) {
        int r = e >> 3, c8 = (e & 7) * 8;
        cp16(&Kw[kbase + (size_t)((kt + 1) * 64 + r) * DK + c8], &nxt[r * 72 + c8]);
      }
    }
    _Float16* Ks = cur ? KV1 : KV0;

    v8f acc = {};
#pragma unroll
    for (int ks = 0; ks < 2; ++ks) {
      union { v16h v; unsigned u[8]; } a, bf;
#pragma unroll
      for (int q = 0; q < 4; ++q) {
        int kk = ks * 32 + 2 * q + hi * 8;
        a.u[q]     = *(const unsigned*)&Qs[mrow * 72 + kk];
        a.u[q + 4] = *(const unsigned*)&Qs[mrow * 72 + kk + 16];
      }
      const int ncol = wn * 16 + ln;   // B[d][n] = K[n][d], contiguous in d
#pragma unroll
      for (int q = 0; q < 8; ++q) {
        int kk = ks * 32 + 2 * (q & 3) + hi * 8 + (q >> 2) * 16;
        bf.u[q] = *(const unsigned*)&Ks[ncol * 72 + kk];
      }
      acc = __builtin_amdgcn_wmma_f32_16x16x32_f16(
          false, a.v, false, bf.v, (short)0, acc, false, false);
    }
#pragma unroll
    for (int r = 0; r < 8; ++r) {
      int row = wq * 16 + r + hi * 8;
      int col = kt * 64 + wn * 16 + ln;
      int dist = col - (q0 + row);
      dist = dist < -32 ? -32 : (dist > 32 ? 32 : dist);
      S[row * 1028 + col] = (acc[r] + qrel[row * 66 + dist + 32]) * 0.125f;
    }
  }
  __syncthreads();

  // --- softmax over full 1024 cols (8 threads per row) ----------------------
  {
    int row = tid >> 3, part = tid & 7;
    float mx = -3.4e38f;
    for (int c = part * 128; c < part * 128 + 128; ++c)
      mx = fmaxf(mx, S[row * 1028 + c]);
    red[row * 8 + part] = mx;
  }
  __syncthreads();
  if (tid < 32) {
    float mx = red[tid * 8];
    for (int p = 1; p < 8; ++p) mx = fmaxf(mx, red[tid * 8 + p]);
    rmax[tid] = mx;
  }
  __syncthreads();
  {
    int row = tid >> 3, part = tid & 7;
    float mx = rmax[row], sum = 0.f;
    for (int c = part * 128; c < part * 128 + 128; ++c) {
      float p = __expf(S[row * 1028 + c] - mx);
      Ps[row * 1032 + c] = (_Float16)p;
      sum += p;
    }
    red[row * 8 + part] = sum;
  }
  __syncthreads();
  if (tid < 32) {
    float s = 0.f;
    for (int p = 0; p < 8; ++p) s += red[tid * 8 + p];
    rinv[tid] = 1.f / s;
  }
  __syncthreads();

  // --- bucket sums for rel-v: idx=clip(k-q) -> 0:prefix, 64:suffix, else 1:1
  {
    int row = tid >> 3, part = tid & 7;
    int qg = q0 + row;
    float b0 = 0.f, b64 = 0.f;
    for (int c = part * 128; c < part * 128 + 128; ++c) {
      int dist = c - qg;
      float p = (float)Ps[row * 1032 + c];
      if (dist <= -32)      b0  += p;
      else if (dist >= 32)  b64 += p;
      else buck[row * 66 + dist + 32] = p;   // unique writer per (row,dist)
    }
    red[row * 8 + part]   = b0;
    qrel[row * 66 + part] = b64;             // qrel is dead now, reuse
  }
  __syncthreads();
  if (tid < 32) {
    float b0 = 0.f, b64 = 0.f;
    for (int p = 0; p < 8; ++p) { b0 += red[tid * 8 + p]; b64 += qrel[tid * 66 + p]; }
    buck[tid * 66 + 0]  = b0;
    buck[tid * 66 + 64] = b64;
  }

  // --- pass 2: O = P V, double-buffered V (already [d][s] in memory) --------
  for (int e = tid; e < 64 * 8; e += 256) {   // V tile 0 -> KV0
    int d = e >> 3, c8 = (e & 7) * 8;
    cp16(&Vt[vbase + (size_t)d * SEQ + c8], &KV0[d * 72 + c8]);
  }
  v8f oacc = {};
#pragma unroll 2
  for (int kt = 0; kt < 16; ++kt) {
    const int cur = kt & 1;
    wait_async();
    __syncthreads();  // KV[cur] ready; buckets visible at kt=0
    if (kt + 1 < 16) {
      _Float16* nxt = cur ? KV0 : KV1;
      for (int e = tid; e < 64 * 8; e += 256) {
        int d = e >> 3, c8 = (e & 7) * 8;
        cp16(&Vt[vbase + (size_t)d * SEQ + (kt + 1) * 64 + c8], &nxt[d * 72 + c8]);
      }
    }
    _Float16* Vs = cur ? KV1 : KV0;
#pragma unroll
    for (int ks = 0; ks < 2; ++ks) {
      union { v16h v; unsigned u[8]; } a, bf;
#pragma unroll
      for (int q = 0; q < 4; ++q) {
        int kk = kt * 64 + ks * 32 + 2 * q + hi * 8;
        a.u[q]     = *(const unsigned*)&Ps[mrow * 1032 + kk];
        a.u[q + 4] = *(const unsigned*)&Ps[mrow * 1032 + kk + 16];
      }
      const int ncol = wn * 16 + ln;                 // output dim d
#pragma unroll
      for (int q = 0; q < 8; ++q) {
        int kk = ks * 32 + 2 * (q & 3) + hi * 8 + (q >> 2) * 16;
        bf.u[q] = *(const unsigned*)&Vs[ncol * 72 + kk];
      }
      oacc = __builtin_amdgcn_wmma_f32_16x16x32_f16(
          false, a.v, false, bf.v, (short)0, oacc, false, false);
    }
  }
  __syncthreads();

  // --- epilogue: add bucket @ relv_table, normalize, store ctx [B,S,D] ------
  const int hidx = bh & (NH - 1), bidx = bh >> 4;
#pragma unroll
  for (int r = 0; r < 8; ++r) {
    int row  = wq * 16 + r + hi * 8;
    int dcol = wn * 16 + ln;
    float o = oacc[r];
    for (int rr = 0; rr < NREL; ++rr) o += buck[row * 66 + rr] * tv[rr * 64 + dcol];
    o *= rinv[row];
    int qg = q0 + row;
    ctx[((size_t)(bidx * SEQ + qg)) * DMODEL + hidx * DK + dcol] = (_Float16)o;
  }
}

// ---------------------------------------------------------------------------
extern "C" void kernel_launch(void* const* d_in, const int* in_sizes, int n_in,
                              void* d_out, int out_size, void* d_ws, size_t ws_size,
                              hipStream_t stream) {
  const float* query = (const float*)d_in[0];
  const float* key   = (const float*)d_in[1];
  const float* value = (const float*)d_in[2];
  const float* Wq = (const float*)d_in[3];  const float* bq = (const float*)d_in[4];
  const float* Wk = (const float*)d_in[5];  const float* bk = (const float*)d_in[6];
  const float* Wv = (const float*)d_in[7];  const float* bv = (const float*)d_in[8];
  const float* Wo = (const float*)d_in[9];  const float* bo = (const float*)d_in[10];
  const float* relk = (const float*)d_in[11];
  const float* relv = (const float*)d_in[12];
  float* out = (float*)d_out;

  const int M = NBATCH * SEQ, N = DMODEL, K = DMODEL;
  const size_t seg = (size_t)16 << 20;   // 16 MB per f16 tensor
  char* ws = (char*)d_ws;
  _Float16* Qw  = (_Float16*)(ws + 0 * seg);
  _Float16* Kw  = (_Float16*)(ws + 1 * seg);
  _Float16* Vw  = (_Float16*)(ws + 2 * seg);   // [B,H,dk,S]
  _Float16* ctx = (_Float16*)(ws + 3 * seg);

  dim3 gblk(256), ggrid(N / 128, M / 128);

  // projections -> f16
  gemm_wt<float, 0><<<ggrid, gblk, 0, stream>>>(query, Wq, bq, (void*)Qw, M, N, K);
  gemm_wt<float, 0><<<ggrid, gblk, 0, stream>>>(key,   Wk, bk, (void*)Kw, M, N, K);
  gemm_wt<float, 2><<<ggrid, gblk, 0, stream>>>(value, Wv, bv, (void*)Vw, M, N, K);

  // attention
  (void)hipFuncSetAttribute((const void*)attn_kernel,
                            hipFuncAttributeMaxDynamicSharedMemorySize,
                            (int)ATTN_SMEM_BYTES);
  attn_kernel<<<dim3(SEQ / 32, NBATCH * NH), dim3(256), ATTN_SMEM_BYTES, stream>>>(
      Qw, Kw, Vw, relk, relv, ctx);

  // output projection -> f32 [B,S,D]
  gemm_wt<_Float16, 1><<<ggrid, gblk, 0, stream>>>(ctx, Wo, bo, (void*)out, M, N, K);
}
